// LocalAtten_16913581211835
// MI455X (gfx1250) — matrix operand order repaired and checked by reading, not compiled, not run on hardware
//
#include <hip/hip_runtime.h>
#include <hip/hip_bf16.h>

#define NI   64
#define ND   64
#define NH   4
#define NB   4
#define NLOC 2048
#define NNEI 64
#define NT   (NB * NLOC)   // 8192 tokens

typedef __attribute__((ext_vector_type(16))) __bf16 v16bf;
typedef __attribute__((ext_vector_type(8)))  float  v8f;
typedef __attribute__((ext_vector_type(4)))  float  fx4;

union V16 {
    v16bf v;
    __bf16 u[16];
    uint4 q[2];
};

// hardware f32 -> bf16 convert (v_cvt_*_bf16_f32 on gfx1250, RNE)
__device__ __forceinline__ __bf16 to_bf(float x) { return (__bf16)x; }

// ---------------------------------------------------------------------------
// P0: fold weights.
//   Mcm[(h*64+i)*64+ip] = sum_d Wq[ip,4d+h] * Wkv[i,8d+h]      (B for scores)
//   Ncm[o*256 + h*64+i] = sum_d Wkv[i,8d+h+4] * Wh[h*64+d, o]  (B for output)
// Both stored column-major (column = WMMA N, contiguous K) in bf16.
// ---------------------------------------------------------------------------
__global__ void prep_kernel(const float* __restrict__ Wq,
                            const float* __restrict__ Wkv,
                            const float* __restrict__ Wh,
                            __bf16* __restrict__ Mcm,
                            __bf16* __restrict__ Ncm) {
    const int tid = threadIdx.x;
    for (int idx = tid; idx < NH * NI * NI; idx += 256) {
        int ip = idx & 63;
        int i  = (idx >> 6) & 63;
        int h  = idx >> 12;
        float s = 0.f;
        for (int d = 0; d < ND; ++d)
            s += Wq[ip * 256 + d * 4 + h] * Wkv[i * 512 + d * 8 + h];
        Mcm[idx] = to_bf(s);
    }
    for (int idx = tid; idx < 64 * 256; idx += 256) {
        int k = idx & 255;
        int o = idx >> 8;
        int i = k & 63;
        int h = k >> 6;
        float s = 0.f;
        for (int d = 0; d < ND; ++d)
            s += Wkv[i * 512 + d * 8 + h + 4] * Wh[(h * 64 + d) * 64 + o];
        Ncm[idx] = to_bf(s);
    }
}

// ---------------------------------------------------------------------------
// P1: qW[t, h, i] = sum_{i'} g1[t, i'] * Mcm[h][i', i]
// One wave per (head, 16-token tile). 4 N-tiles x 2 K-steps of wmma bf16.
// ---------------------------------------------------------------------------
__global__ void qw_kernel(const float* __restrict__ g1,
                          const __bf16* __restrict__ Mcm,
                          __bf16* __restrict__ qW) {
    const int lane = threadIdx.x & 31;
    const int wave = threadIdx.x >> 5;
    const int task = blockIdx.x * 8 + wave;   // 2048 = 4 heads * 512 tiles
    const int h    = task & 3;
    const int t0   = (task >> 2) * 16;
    const int mrow = lane & 15;
    const int koA  = (lane < 16) ? 0 : 8;     // A K sub-offset per ISA layout
    const int koB  = (lane < 16) ? 0 : 8;     // B K sub-offset per ISA layout

    v8f acc[4] = {};
    for (int kb = 0; kb < 2; ++kb) {
        // A: g1 row (t0+mrow), K elems {koA..koA+7, koA+16..koA+23} + 32*kb
        const float* rp = g1 + (size_t)(t0 + mrow) * NI + kb * 32 + koA;
        float fb[16];
        *reinterpret_cast<fx4*>(&fb[0])  = *reinterpret_cast<const fx4*>(rp);
        *reinterpret_cast<fx4*>(&fb[4])  = *reinterpret_cast<const fx4*>(rp + 4);
        *reinterpret_cast<fx4*>(&fb[8])  = *reinterpret_cast<const fx4*>(rp + 16);
        *reinterpret_cast<fx4*>(&fb[12]) = *reinterpret_cast<const fx4*>(rp + 20);
        V16 a;
#pragma unroll
        for (int e = 0; e < 16; ++e) a.u[e] = to_bf(fb[e]);
#pragma unroll
        for (int nt = 0; nt < 4; ++nt) {
            const __bf16* bp =
                Mcm + (size_t)(h * 64 + nt * 16 + mrow) * 64 + kb * 32 + koB;
            V16 b;
            b.q[0] = *reinterpret_cast<const uint4*>(bp);
            b.q[1] = *reinterpret_cast<const uint4*>(bp + 16);
            acc[nt] = __builtin_amdgcn_wmma_f32_16x16x32_bf16(
                false, a.v, false, b.v, (short)0, acc[nt], false, false);
        }
    }
#pragma unroll
    for (int nt = 0; nt < 4; ++nt)
#pragma unroll
        for (int r = 0; r < 8; ++r) {
            int tok = t0 + r + ((lane < 16) ? 0 : 8);
            qW[((size_t)tok * 4 + h) * 64 + nt * 16 + mrow] = to_bf(acc[nt][r]);
        }
}

// ---------------------------------------------------------------------------
// P2: per-token fused attention. gg1 tile read exactly once (HBM-bound part),
// streamed with non-temporal loads so it does not evict reused data from L2.
//   scores[h,j] = (qW[t,h,:] . gg1[t,j,:]) / 8   -> masked softmax over j
//   Aflat[t, h*64+i] = sum_j w[h,j] * gg1[t,j,i]
// ---------------------------------------------------------------------------
__global__ void atten_kernel(const float* __restrict__ gg1,
                             const unsigned char* __restrict__ mask,
                             const __bf16* __restrict__ qW,
                             __bf16* __restrict__ Aflat) {
    __shared__ __align__(16) __bf16 gg_rm[64][64]; // [j][i]
    __shared__ __align__(16) __bf16 gg_cm[64][64]; // [i][j]
    __shared__ __align__(16) __bf16 s_w[16][64];   // padded attn-weight A tile
    __shared__ float s_sc[4][64];
    __shared__ float s_red[256];

    const int tid  = threadIdx.x;
    const int lane = tid & 31;
    const int t    = blockIdx.x;

    // ---- stream gg1 tile (64x64 f32) -> bf16 in both orientations
    const fx4* G4 = reinterpret_cast<const fx4*>(gg1 + (size_t)t * (NNEI * NI));
#pragma unroll
    for (int kq = 0; kq < 4; ++kq) {
        int q4 = tid + kq * 256;            // 1024 float4s
        fx4 f = __builtin_nontemporal_load(G4 + q4);
        int j  = q4 >> 4;
        int i0 = (q4 & 15) * 4;
        __bf16 u0 = to_bf(f.x), u1 = to_bf(f.y), u2 = to_bf(f.z), u3 = to_bf(f.w);
        gg_rm[j][i0 + 0] = u0; gg_rm[j][i0 + 1] = u1;
        gg_rm[j][i0 + 2] = u2; gg_rm[j][i0 + 3] = u3;
        gg_cm[i0 + 0][j] = u0; gg_cm[i0 + 1][j] = u1;
        gg_cm[i0 + 2][j] = u2; gg_cm[i0 + 3][j] = u3;
    }
    for (int e = tid; e < 16 * 64; e += 256) s_w[e >> 6][e & 63] = (__bf16)0.0f;
    __syncthreads();

    // ---- scores = qW (4 rows, zero-padded to 16) @ gg1^T   (wave 0)
    if (tid < 32) {
        const int mrow = lane & 15;
        const int koA  = (lane < 16) ? 0 : 8;
        const int koB  = (lane < 16) ? 0 : 8;
        v8f acc[4] = {};
        const __bf16* qwt = qW + (size_t)t * 256;
        for (int kb = 0; kb < 2; ++kb) {
            V16 a;
            if (mrow < 4) {
                const __bf16* rp = qwt + mrow * 64 + kb * 32 + koA;
                a.q[0] = *reinterpret_cast<const uint4*>(rp);
                a.q[1] = *reinterpret_cast<const uint4*>(rp + 16);
            } else {
                a.q[0] = make_uint4(0, 0, 0, 0);
                a.q[1] = make_uint4(0, 0, 0, 0);
            }
#pragma unroll
            for (int nt = 0; nt < 4; ++nt) {
                const __bf16* bp = &gg_rm[nt * 16 + mrow][kb * 32 + koB];
                V16 b;
                b.q[0] = *reinterpret_cast<const uint4*>(bp);
                b.q[1] = *reinterpret_cast<const uint4*>(bp + 16);
                acc[nt] = __builtin_amdgcn_wmma_f32_16x16x32_bf16(
                    false, a.v, false, b.v, (short)0, acc[nt], false, false);
            }
        }
        if (lane < 16) {
#pragma unroll
            for (int nt = 0; nt < 4; ++nt)
#pragma unroll
                for (int r = 0; r < 4; ++r)
                    s_sc[r][nt * 16 + lane] = acc[nt][r] * 0.125f; // 1/sqrt(64)
        }
    }
    __syncthreads();

    // ---- masked softmax over j, per head (256 threads: h=tid/64, j=tid%64)
    const int h = tid >> 6;
    const int j = tid & 63;
    const unsigned char mv = mask[(size_t)t * NNEI + j];
    float x = mv ? s_sc[h][j] : -__builtin_inff();
    s_red[tid] = x;
    for (int s = 32; s > 0; s >>= 1) {
        __syncthreads();
        if ((tid & 63) < s) {
            float o = s_red[tid + s];
            if (o > s_red[tid]) s_red[tid] = o;
        }
    }
    __syncthreads();
    float mx = s_red[tid & ~63];
    __syncthreads();
    float e = (mv && mx > -3.0e38f) ? __expf(x - mx) : 0.f;
    s_red[tid] = e;
    for (int s = 32; s > 0; s >>= 1) {
        __syncthreads();
        if ((tid & 63) < s) s_red[tid] += s_red[tid + s];
    }
    __syncthreads();
    float sum = s_red[tid & ~63];
    float w = (sum > 0.f) ? (e / sum) : 0.f;
    s_w[h][j] = to_bf(w);
    __syncthreads();

    // ---- Aflat[t] = w (4x64, padded) @ gg1 (64x64)   (wave 0)
    if (tid < 32) {
        const int mrow = lane & 15;
        const int koA  = (lane < 16) ? 0 : 8;
        const int koB  = (lane < 16) ? 0 : 8;
        v8f acc[4] = {};
        for (int kb = 0; kb < 2; ++kb) {
            const __bf16* rp = &s_w[mrow][kb * 32 + koA];
            V16 a;
            a.q[0] = *reinterpret_cast<const uint4*>(rp);
            a.q[1] = *reinterpret_cast<const uint4*>(rp + 16);
#pragma unroll
            for (int nt = 0; nt < 4; ++nt) {
                const __bf16* bp = &gg_cm[nt * 16 + mrow][kb * 32 + koB];
                V16 b;
                b.q[0] = *reinterpret_cast<const uint4*>(bp);
                b.q[1] = *reinterpret_cast<const uint4*>(bp + 16);
                acc[nt] = __builtin_amdgcn_wmma_f32_16x16x32_bf16(
                    false, a.v, false, b.v, (short)0, acc[nt], false, false);
            }
        }
        if (lane < 16) {
            __bf16* out = Aflat + (size_t)t * 256;
#pragma unroll
            for (int nt = 0; nt < 4; ++nt)
#pragma unroll
                for (int r = 0; r < 4; ++r)
                    out[r * 64 + nt * 16 + lane] = to_bf(acc[nt][r]);
        }
    }
}

// ---------------------------------------------------------------------------
// P3: out = Aflat (8192x256) @ Ncm (256x64) + bh.  One wave per 16-token tile.
// ---------------------------------------------------------------------------
__global__ void out_kernel(const __bf16* __restrict__ Aflat,
                           const __bf16* __restrict__ Ncm,
                           const float* __restrict__ bh,
                           float* __restrict__ out) {
    const int lane = threadIdx.x & 31;
    const int wave = threadIdx.x >> 5;
    const int t0   = (blockIdx.x * 8 + wave) * 16;   // 512 token tiles
    const int mrow = lane & 15;
    const int koA  = (lane < 16) ? 0 : 8;
    const int koB  = (lane < 16) ? 0 : 8;

    v8f acc[4] = {};
    for (int kb = 0; kb < 8; ++kb) {
        const __bf16* rp = Aflat + (size_t)(t0 + mrow) * 256 + kb * 32 + koA;
        V16 a;
        a.q[0] = *reinterpret_cast<const uint4*>(rp);
        a.q[1] = *reinterpret_cast<const uint4*>(rp + 16);
#pragma unroll
        for (int nt = 0; nt < 4; ++nt) {
            const __bf16* bp = Ncm + (size_t)(nt * 16 + mrow) * 256 + kb * 32 + koB;
            V16 b;
            b.q[0] = *reinterpret_cast<const uint4*>(bp);
            b.q[1] = *reinterpret_cast<const uint4*>(bp + 16);
            acc[nt] = __builtin_amdgcn_wmma_f32_16x16x32_bf16(
                false, a.v, false, b.v, (short)0, acc[nt], false, false);
        }
    }
#pragma unroll
    for (int nt = 0; nt < 4; ++nt) {
        int o = nt * 16 + mrow;
        float bias = bh[o];
#pragma unroll
        for (int r = 0; r < 8; ++r) {
            int tok = t0 + r + ((lane < 16) ? 0 : 8);
            out[(size_t)tok * 64 + o] = acc[nt][r] + bias;
        }
    }
}

extern "C" void kernel_launch(void* const* d_in, const int* in_sizes, int n_in,
                              void* d_out, int out_size, void* d_ws, size_t ws_size,
                              hipStream_t stream) {
    (void)in_sizes; (void)n_in; (void)out_size; (void)ws_size;
    const float* g1  = (const float*)d_in[0];
    const float* gg1 = (const float*)d_in[1];
    const unsigned char* mask = (const unsigned char*)d_in[2]; // jax bool = 1 byte
    const float* Wq  = (const float*)d_in[3];
    const float* Wkv = (const float*)d_in[4];
    const float* Wh  = (const float*)d_in[5];
    const float* bh  = (const float*)d_in[6];
    float* out = (float*)d_out;

    char* ws = (char*)d_ws;
    __bf16* Mcm = (__bf16*)(ws);                                  // 32 KB
    __bf16* Ncm = (__bf16*)(ws + 32768);                          // 32 KB
    __bf16* qW  = (__bf16*)(ws + 65536);                          // 4 MB
    __bf16* Af  = (__bf16*)(ws + 65536 + (size_t)NT * 256 * 2);   // 4 MB

    prep_kernel<<<1, 256, 0, stream>>>(Wq, Wkv, Wh, Mcm, Ncm);
    qw_kernel<<<256, 256, 0, stream>>>(g1, Mcm, qW);
    atten_kernel<<<NT, 256, 0, stream>>>(gg1, mask, qW, Af);
    out_kernel<<<64, 256, 0, stream>>>(Af, Ncm, bh, out);
}
